// SOHMM_60988535603242
// MI455X (gfx1250) — matrix operand order repaired and checked by reading, not compiled, not run on hardware
//
#include <hip/hip_runtime.h>
#include <hip/hip_bf16.h>

typedef __bf16 bf16_t;
typedef __attribute__((ext_vector_type(16))) __bf16 v16bf;
typedef __attribute__((ext_vector_type(8)))  float  v8f;

#define HH 64
#define VV 32000
#define BB 32
#define TT 512
#define BT 16   // batch columns per block

// ---- prologue: repack alpha_exp (f32) into per-lane WMMA A-tile layout (bf16) ----
// Tile (j, it, ks) = A 16x32 bf16 operand for rows i=it*16..+15, K=ks*32..+31.
// Per ISA layout: lane L (half = L>>4, m = L&15): row = it*16 + m,
//   elements 0..7  -> K = ks*32 + half*8 + 0..7
//   elements 8..15 -> K = ks*32 + half*8 + 16..23
// Packed: [j][it][ks][lane][16 bf16]  => wave load = 32 lanes x 32B contiguous.
__global__ __launch_bounds__(512) void pack_alpha_kernel(const float* __restrict__ a,
                                                         bf16_t* __restrict__ o) {
  int idx = blockIdx.x * blockDim.x + threadIdx.x;   // 0 .. 262143
  int q    = idx & 15;
  int lane = (idx >> 4) & 31;
  int tile = idx >> 9;
  int ks   = tile & 1;
  int it   = (tile >> 1) & 3;
  int j    = tile >> 3;
  int m    = lane & 15;
  int half = lane >> 4;
  int row  = it * 16 + m;
  int k    = ks * 32 + half * 8 + ((q < 8) ? q : (q + 8));
  o[idx] = (bf16_t)a[((size_t)row * HH + j) * HH + k];
}

// ---- main recurrence: one block owns 16 batch columns for all T steps ----
__global__ __launch_bounds__(1024) void sohmm_kernel(
    const bf16_t* __restrict__ alpha_pk,  // packed tiles, 512 KB
    const float*  __restrict__ beta,      // [H][V]
    const float*  __restrict__ gamma,     // [H][H]
    const int*    __restrict__ ids,       // [B][T]
    float*        __restrict__ out)       // ys [T][H][H][B] then final [B]
{
  __shared__ float yLDS[HH * HH * BT];    // 256 KB, layout (a*64+c)*16 + b
  __shared__ float ipLDS[HH * BT];        // 4 KB emission for current step
  __shared__ float redLDS[HH * BT];       // 4 KB reduction scratch
  __shared__ float ymaxLDS[BT];
  __shared__ float scal[2];

  const int tid   = threadIdx.x;
  const int lane  = tid & 31;
  const int wid   = tid >> 5;             // 0..31 (32 wave32 waves)
  const int bbase = blockIdx.x * BT;
  const int jb_j  = tid >> 4;             // 0..63 (row index for scalar phases)
  const int jb_b  = tid & 15;             // 0..15 (batch column)
  const int nc    = lane & 15;            // WMMA N column == batch column
  const int half  = lane >> 4;            // 0/1 lane half
  const int j0    = wid * 2;              // this wave owns j0, j0+1

  // Per-lane packed-A pointer: tile stride = 32 lanes * 16 elems
  const v16bf* apk = (const v16bf*)alpha_pk + lane;   // index by tile*32

  // ---- init: y0[i,j,b] = ip[T-1][j,b]; ys[0] = y0 ----
  {
    int tok  = ids[(bbase + jb_b) * TT + (TT - 1)];
    float ip = (tok >= 0) ? beta[jb_j * VV + tok] : 0.0f;
    ipLDS[jb_j * BT + jb_b] = ip;
    for (int i = 0; i < HH; ++i) {
      yLDS[(i * HH + jb_j) * BT + jb_b] = ip;
      out[(size_t)(i * HH + jb_j) * BB + bbase + jb_b] = ip;
    }
  }
  __syncthreads();

  for (int s = 1; s < TT; ++s) {
    const int te = TT - 1 - s;

    // ---- Phase A: emission gather + per-b max (partial) ----
    {
      int tok = ids[(bbase + jb_b) * TT + te];
      ipLDS[jb_j * BT + jb_b] = (tok >= 0) ? beta[jb_j * VV + tok] : 0.0f;
      float m = -3.0e38f;
      const int base = jb_j * HH * BT + jb_b;
      #pragma unroll 4
      for (int c = 0; c < HH; ++c) m = fmaxf(m, yLDS[base + c * BT]);
      redLDS[jb_j * BT + jb_b] = m;
    }
    __syncthreads();
    if (tid < BT) {
      float m = -3.0e38f;
      for (int c = 0; c < HH; ++c) m = fmaxf(m, redLDS[c * BT + tid]);
      ymaxLDS[tid] = m;
    }
    __syncthreads();

    // ---- Phase B: prefetch A tiles, stage bf16 B-operands exp(y-ymax) ----
    {
      // Warm this wave's 16 packed A tiles (1 KB each) while doing exp math.
      #pragma unroll 1
      for (int t = 0; t < 16; ++t)
        __builtin_prefetch((const void*)(apk + (j0 * 8 + t) * 32), 0, 3);
    }
    const float ymB = ymaxLDS[nc];
    auto make_B = [&](int j, int ks) -> v16bf {
      v16bf bb;
      #pragma unroll
      for (int p = 0; p < 8; ++p) {
        const int k = ks * 32 + 2 * p + half * 8 + ((p >= 4) ? 8 : 0);
        float e0 = __expf(yLDS[(j * HH + k    ) * BT + nc] - ymB);
        float e1 = __expf(yLDS[(j * HH + k + 1) * BT + nc] - ymB);
        bb[2 * p]     = (bf16_t)e0;
        bb[2 * p + 1] = (bf16_t)e1;
      }
      return bb;
    };
    v16bf Bst00 = make_B(j0,     0);
    v16bf Bst01 = make_B(j0,     1);
    v16bf Bst10 = make_B(j0 + 1, 0);
    v16bf Bst11 = make_B(j0 + 1, 1);
    __syncthreads();   // all reads of old y complete -> safe to overwrite in place

    // ---- Phase C: WMMA accumulate over K, then log + ymax + emission ----
    {
      const float ym = ymaxLDS[nc];
      // Rolled tile loop, compile-time-constant B operands: no spills.
      auto do_slice = [&](int j, const v16bf& B0, const v16bf& B1) {
        const float ip = ipLDS[j * BT + nc];
        #pragma unroll 1
        for (int it = 0; it < 4; ++it) {
          const int tbase = (j * 4 + it) * 2;
          v16bf a0 = apk[(tbase    ) * 32];   // coalesced 32B/lane
          v16bf a1 = apk[(tbase + 1) * 32];
          v8f acc = {};
          acc = __builtin_amdgcn_wmma_f32_16x16x32_bf16(
                  false, a0, false, B0, (short)0, acc, false, false);
          acc = __builtin_amdgcn_wmma_f32_16x16x32_bf16(
                  false, a1, false, B1, (short)0, acc, false, false);
          // C/D layout: VGPR r -> M = r + 8*half, N = lane&15
          #pragma unroll
          for (int r = 0; r < 8; ++r) {
            const int i = it * 16 + r + half * 8;
            float v = __logf(acc[r]) + ym + ip;
            yLDS[(i * HH + j) * BT + nc] = v;
            out[((size_t)s * HH * HH + (size_t)i * HH + j) * BB + bbase + nc] = v;
          }
        }
      };
      do_slice(j0,     Bst00, Bst01);
      do_slice(j0 + 1, Bst10, Bst11);
    }
    __syncthreads();
  }

  // ---- epilogue: final[b] = log( sum_ij softmax(gamma)[i,j] * exp(y-ymax) ) + ymax ----
  {
    float m = -3.0e38f;
    const int base = jb_j * HH * BT + jb_b;
    for (int c = 0; c < HH; ++c) m = fmaxf(m, yLDS[base + c * BT]);
    redLDS[jb_j * BT + jb_b] = m;
  }
  __syncthreads();
  if (tid < BT) {
    float m = -3.0e38f;
    for (int c = 0; c < HH; ++c) m = fmaxf(m, redLDS[c * BT + tid]);
    ymaxLDS[tid] = m;
  }
  __syncthreads();
  {  // gamma flat max
    float m = fmaxf(fmaxf(gamma[tid*4], gamma[tid*4+1]),
                    fmaxf(gamma[tid*4+2], gamma[tid*4+3]));
    redLDS[tid] = m;
  }
  __syncthreads();
  if (tid == 0) {
    float m = -3.0e38f;
    for (int c = 0; c < 1024; ++c) m = fmaxf(m, redLDS[c]);
    scal[0] = m;
  }
  __syncthreads();
  {  // gamma flat sum of exp
    const float gm = scal[0];
    float ssum = 0.0f;
    for (int q = 0; q < 4; ++q) ssum += __expf(gamma[tid*4+q] - gm);
    redLDS[tid] = ssum;
  }
  __syncthreads();
  if (tid == 0) {
    float ssum = 0.0f;
    for (int c = 0; c < 1024; ++c) ssum += redLDS[c];
    scal[1] = ssum;
  }
  __syncthreads();
  {  // weighted sum per batch column
    const float gm = scal[0], ym = ymaxLDS[jb_b];
    float ssum = 0.0f;
    for (int c = 0; c < HH; ++c) {
      const int e = jb_j * HH + c;  // i = jb_j, j = c
      ssum += __expf(gamma[e] - gm) * __expf(yLDS[e * BT + jb_b] - ym);
    }
    redLDS[jb_j * BT + jb_b] = ssum;
  }
  __syncthreads();
  if (tid < BT) {
    float ssum = 0.0f;
    for (int c = 0; c < HH; ++c) ssum += redLDS[c * BT + tid];
    out[(size_t)TT * HH * HH * BB + bbase + tid] =
        __logf(ssum) - __logf(scal[1]) + ymaxLDS[tid];
  }
}

extern "C" void kernel_launch(void* const* d_in, const int* in_sizes, int n_in,
                              void* d_out, int out_size, void* d_ws, size_t ws_size,
                              hipStream_t stream) {
  const float* alpha = (const float*)d_in[0];   // [H,H,H] f32
  const float* beta  = (const float*)d_in[1];   // [H,V]   f32
  const float* gamma = (const float*)d_in[2];   // [H,H]   f32
  const int*   ids   = (const int*)d_in[3];     // [B,T]   int
  float* out = (float*)d_out;
  bf16_t* alpha_pk = (bf16_t*)d_ws;             // 512 KB packed bf16 alpha tiles

  const int n = HH * HH * HH;                   // 262144 packed elements
  pack_alpha_kernel<<<n / 512, 512, 0, stream>>>(alpha, alpha_pk);
  sohmm_kernel<<<BB / BT, 1024, 0, stream>>>(alpha_pk, beta, gamma, ids, out);
}
